// HFBlock_73538430042706
// MI455X (gfx1250) — compile-verified, matrix operand order checked by reading, not run on hardware
//
#include <hip/hip_runtime.h>
#include <math.h>

// ---------------- problem constants ----------------
#define BB   4
#define CC   96
#define HH   64
#define WW   64
#define LL   4096      // H*W
#define DIN  192
#define NN   16
#define RR   6
#define KK   4
#define EPSF 1e-5f

typedef __attribute__((ext_vector_type(16))) _Float16 v16h;
typedef __attribute__((ext_vector_type(8)))  float    v8f;
typedef __attribute__((ext_vector_type(4)))  unsigned int u32x4;
typedef __attribute__((ext_vector_type(8)))  int          i32x8;
typedef __attribute__((ext_vector_type(4)))  int          i32x4;

#if __has_builtin(__builtin_amdgcn_tensor_load_to_lds)
#define HAVE_TDM 1
#else
#define HAVE_TDM 0
#endif

// ---------------- WMMA helpers (wave32, 16x16x32 f16 -> f32) ----------------
__device__ __forceinline__ v8f wmma_acc(v16h a, v16h b, v8f c) {
  // 8 args: (neg_a, A, neg_b, B, c_mod, C, reuse_a, reuse_b)
  return __builtin_amdgcn_wmma_f32_16x16x32_f16(false, a, false, b, (short)0, c,
                                                false, false);
}

// A-matrix fragment (16xK slice, contiguous k order as in bfrag_g: lanes 0-15 -> M rows with
// K=kk..kk+15, lanes 16-31 -> same rows with K=kk+16..kk+31)
__device__ __forceinline__ v16h afrag(const _Float16* sA, int ldA, int kk, int lane) {
  const _Float16* p = sA + (size_t)(lane & 15) * ldA + kk + ((lane >> 4) << 4);
  return *(const v16h*)p;
}

// B-matrix fragment built from a global f32 weight matrix stored row-major as
// W[out][in] (so B[k][n] = W[n][k]; each lane reads 16 contiguous floats).
__device__ __forceinline__ v16h bfrag_g(const float* __restrict__ w, int ldw,
                                        int row, int kk, int lane) {
  const float* p = w + (size_t)row * ldw + kk + ((lane >> 4) << 4);
  v16h r;
#pragma unroll
  for (int j = 0; j < 16; ++j) r[j] = (_Float16)p[j];
  return r;
}

// GEMM: (16 x K) f16 tile in LDS  x  W^T  ->  f32 tile in LDS.
// W is global f32, row-major W[out][K], 'nrows' valid output rows (clamped).
__device__ __forceinline__ void gemm_lds(const _Float16* sA, int ldA,
                                         const float* __restrict__ W, int ldw,
                                         int nrows, float* sV, int ldv,
                                         int ntiles, int ksteps,
                                         const float* __restrict__ bias, int lane) {
  for (int t = 0; t < ntiles; ++t) {
    int n   = t * 16 + (lane & 15);
    int row = (n < nrows) ? n : (nrows - 1);
    float bv = bias ? bias[row] : 0.f;
    v8f acc;
#pragma unroll
    for (int r = 0; r < 8; ++r) acc[r] = bv;
    for (int ks = 0; ks < ksteps; ++ks) {
      v16h a  = afrag(sA, ldA, ks * 32, lane);
      v16h bf = bfrag_g(W, ldw, row, ks * 32, lane);
      acc = wmma_acc(a, bf, acc);
    }
    int mo = (lane >> 4) * 8;  // C/D layout: vgpr r -> M = r + 8*(lane>=16)
#pragma unroll
    for (int r = 0; r < 8; ++r)
      sV[(size_t)(mo + r) * ldv + t * 16 + (lane & 15)] = acc[r];
  }
}

// ---------------- TDM: 2D tile (f32) global -> LDS ----------------
// D# per CDNA5 ISA ch.8: group0 {count=1, lds_addr, global_addr57, type=2},
// group1 {data_size=4B, tensor_dim0/1, tile_dim0/1, tensor_dim0_stride},
// groups 2/3 zero (tensor <= 2D).
__device__ __forceinline__ void tdm_load_2d_f32(const void* gptr,
                                                unsigned int lds_byte_addr,
                                                unsigned int dim0, unsigned int dim1,
                                                unsigned int tile0, unsigned int tile1,
                                                unsigned int stride0) {
#if HAVE_TDM
  unsigned long long ga = (unsigned long long)gptr;
  u32x4 g0;
  g0[0] = 1u;                                   // count=1, user descriptor
  g0[1] = lds_byte_addr;                        // LDS byte address
  g0[2] = (unsigned int)ga;                     // global_addr[31:0]
  g0[3] = (unsigned int)(ga >> 32) | 0x80000000u; // addr[56:32] | type=2<<30
  i32x8 g1;
  g1[0] = (int)(2u << 16);                      // data_size code 2 = 4 bytes
  g1[1] = (int)((dim0 & 0xFFFFu) << 16);        // tensor_dim0[15:0] @ bits63:48
  g1[2] = (int)(((dim0 >> 16) & 0xFFFFu) | ((dim1 & 0xFFFFu) << 16));
  g1[3] = (int)(((dim1 >> 16) & 0xFFFFu) | ((tile0 & 0xFFFFu) << 16));
  g1[4] = (int)(tile1 & 0xFFFFu);               // tile_dim1 (tile_dim2=0)
  g1[5] = (int)stride0;                         // tensor_dim0_stride[31:0]
  g1[6] = 0;
  g1[7] = 0;
  i32x4 gz = {0, 0, 0, 0};
#if defined(__clang_major__) && (__clang_major__ >= 23)
  i32x8 gz8 = {0, 0, 0, 0, 0, 0, 0, 0};
  __builtin_amdgcn_tensor_load_to_lds(g0, g1, gz, gz, gz8, 0);
#else
  __builtin_amdgcn_tensor_load_to_lds(g0, g1, gz, gz, 0);
#endif
#endif
}

__device__ __forceinline__ void tdm_wait0() {
#if __has_builtin(__builtin_amdgcn_s_wait_tensorcnt)
  __builtin_amdgcn_s_wait_tensorcnt(0);
#else
  asm volatile("s_wait_tensorcnt 0x0" ::: "memory");
#endif
}

// ---------------- small device helpers ----------------
__device__ __forceinline__ void ln_row(float* row, int n,
                                       const float* __restrict__ w,
                                       const float* __restrict__ b) {
  float m = 0.f;
  for (int i = 0; i < n; ++i) m += row[i];
  m /= (float)n;
  float v = 0.f;
  for (int i = 0; i < n; ++i) { float d = row[i] - m; v += d * d; }
  v /= (float)n;
  float inv = rsqrtf(v + EPSF);
  for (int i = 0; i < n; ++i) row[i] = (row[i] - m) * inv * w[i] + b[i];
}

// scan-direction index map: scan position l of direction k <-> row-major pixel
// (identical for the input gather and the output scatter)
__device__ __forceinline__ int dir_pos(int k, int l) {
  int t = (k & 2) ? (LL - 1 - l) : l;
  if (k & 1) t = ((t & 63) << 6) | (t >> 6);  // (h,w) transpose, H==W==64
  return t;
}

// ======================================================================
// Kernel 1: x_sum -> conv1 -> LN -> GELU -> conv2 -> LN -> LN -> in_proj
// block: 128 threads = 4 waves, 64 pixels (16 per wave). grid: 256.
// Stage-0 tiles arrive via the Tensor Data Mover when available.
// ======================================================================
__global__ __launch_bounds__(128) void k_pre(
    const float* __restrict__ xd, const float* __restrict__ xsh,
    const float* __restrict__ w1, const float* __restrict__ b1,
    const float* __restrict__ ln1w, const float* __restrict__ ln1b,
    const float* __restrict__ w2, const float* __restrict__ b2,
    const float* __restrict__ ln2w, const float* __restrict__ ln2b,
    const float* __restrict__ nw, const float* __restrict__ nb,
    const float* __restrict__ wi,
    float* __restrict__ g_xsum, float* __restrict__ g_xm,
    float* __restrict__ g_z) {
  __shared__ alignas(32) _Float16 s_a[64][CC];
  __shared__ alignas(32) float    s_v[64][CC];
#if HAVE_TDM
  __shared__ alignas(16) float    s_xd[CC][64];   // TDM lands tiles [ch][px]
  __shared__ alignas(16) float    s_xs[CC][64];
#endif
  int tid = threadIdx.x, lane = tid & 31, wid = tid >> 5;
  int pix0 = blockIdx.x * 64;
  __builtin_prefetch(w1, 0, 1);
  __builtin_prefetch(wi, 0, 1);

#if HAVE_TDM
  // stage 0a: async 2D tile DMA (96 channel-rows x 64 pixels) into LDS
  if (wid == 0) {
    int b = pix0 >> 12, hw0 = pix0 & 4095;
    const float* pd = xd  + (((size_t)b * CC) << 12) + hw0;
    const float* ps = xsh + (((size_t)b * CC) << 12) + hw0;
    tdm_load_2d_f32(pd, (unsigned int)(size_t)&s_xd[0][0],
                    LL, CC, 64, CC, LL);
    tdm_load_2d_f32(ps, (unsigned int)(size_t)&s_xs[0][0],
                    LL, CC, 64, CC, LL);
    tdm_wait0();
  }
  __syncthreads();
  // stage 0b: sum, stash f32 + f16, write residual copy
  for (int idx = tid; idx < 64 * CC; idx += 128) {
    int ch = idx >> 6, px = idx & 63;
    float v = s_xd[ch][px] + s_xs[ch][px];
    s_v[px][ch] = v;
    s_a[px][ch] = (_Float16)v;
    int p = pix0 + px, b = p >> 12, hw = p & 4095;
    g_xsum[((size_t)(b * CC + ch) << 12) + hw] = v;
  }
#else
  // fallback: cooperative vector loads
  for (int idx = tid; idx < 64 * CC; idx += 128) {
    int ch = idx >> 6, px = idx & 63;
    int p = pix0 + px, b = p >> 12, hw = p & 4095;
    size_t g = ((size_t)(b * CC + ch) << 12) + hw;
    float v = xd[g] + xsh[g];
    s_v[px][ch] = v;
    s_a[px][ch] = (_Float16)v;
    g_xsum[g] = v;
  }
#endif
  __syncthreads();

  // conv1 (96x96) via WMMA
  gemm_lds(&s_a[wid * 16][0], CC, w1, CC, CC, &s_v[wid * 16][0], CC, 6, 3, b1, lane);
  __syncthreads();

  // LN1 + exact GELU
  if (tid < 64) {
    float* row = s_v[tid];
    ln_row(row, CC, ln1w, ln1b);
    for (int i = 0; i < CC; ++i) {
      float x = row[i];
      float gl = 0.5f * x * (1.f + erff(x * 0.70710678118654752f));
      s_a[tid][i] = (_Float16)gl;
    }
  }
  __syncthreads();

  // conv2 (96x96)
  gemm_lds(&s_a[wid * 16][0], CC, w2, CC, CC, &s_v[wid * 16][0], CC, 6, 3, b2, lane);
  __syncthreads();

  // LN2 then norm
  if (tid < 64) {
    float* row = s_v[tid];
    ln_row(row, CC, ln2w, ln2b);
    ln_row(row, CC, nw, nb);
    for (int i = 0; i < CC; ++i) s_a[tid][i] = (_Float16)row[i];
  }
  __syncthreads();

  // in_proj (384x96): 24 N-tiles, store straight to global xm / z
  for (int t = 0; t < 24; ++t) {
    v8f acc;
#pragma unroll
    for (int r = 0; r < 8; ++r) acc[r] = 0.f;
    int o = t * 16 + (lane & 15);
    for (int ks = 0; ks < 3; ++ks) {
      v16h a  = afrag(&s_a[wid * 16][0], CC, ks * 32, lane);
      v16h bf = bfrag_g(wi, CC, o, ks * 32, lane);
      acc = wmma_acc(a, bf, acc);
    }
    int dch = (o < DIN) ? o : (o - DIN);
    float* dst = (o < DIN) ? g_xm : g_z;
    int mo = (lane >> 4) * 8;
#pragma unroll
    for (int r = 0; r < 8; ++r) {
      int p = pix0 + wid * 16 + mo + r;
      int b = p >> 12, hw = p & 4095;
      dst[((size_t)(b * DIN + dch) << 12) + hw] = acc[r];
    }
  }
}

// ======================================================================
// Kernel 2: depthwise 3x3 conv (pad 1) + SiLU
// ======================================================================
__global__ void k_dwconv(const float* __restrict__ g_xm,
                         const float* __restrict__ wdw,
                         const float* __restrict__ bdw,
                         float* __restrict__ g_xa) {
  int idx = blockIdx.x * 256 + threadIdx.x;       // [0, B*DIN*L)
  if (idx >= BB * DIN * LL) return;
  int hw = idx & 4095;
  int bd = idx >> 12;
  int d  = bd % DIN;
  int h = hw >> 6, w = hw & 63;
  float acc = bdw[d];
#pragma unroll
  for (int i = 0; i < 3; ++i)
#pragma unroll
    for (int j = 0; j < 3; ++j) {
      int hh = h + i - 1, ww = w + j - 1;
      if (hh >= 0 && hh < HH && ww >= 0 && ww < WW)
        acc += wdw[d * 9 + i * 3 + j] * g_xm[((size_t)bd << 12) + (hh << 6) + ww];
    }
  g_xa[idx] = acc / (1.f + __expf(-acc));         // SiLU
}

// ======================================================================
// Kernel 3: per (b,k,l-tile): gather direction, x_proj via WMMA (38 rows,
// padded to 48), then dt expansion + softplus, split B/C.
// block: 128 thr / 4 waves / 64 scan-positions. grid: B*K*64 = 1024.
// ======================================================================
__global__ __launch_bounds__(128) void k_proj(
    const float* __restrict__ g_xa, const float* __restrict__ xpw,
    const float* __restrict__ dtw, const float* __restrict__ dtb,
    float* __restrict__ g_xst, float* __restrict__ g_dts,
    float* __restrict__ g_Bs, float* __restrict__ g_Cs) {
  __shared__ alignas(32) _Float16 s_a[64][DIN];
  __shared__ alignas(32) float    s_v[64][48];
  int tid = threadIdx.x, lane = tid & 31, wid = tid >> 5;
  int blk = blockIdx.x;
  int lt = blk & 63, bk = blk >> 6;
  int k = bk & 3, b = bk >> 2;
  int l0 = lt * 64;

  // gather xs tile, stage f16 for WMMA, dump f32 in scan layout (bk,l,d)
  for (int idx = tid; idx < 64 * DIN; idx += 128) {
    int d = idx % DIN, px = idx / DIN;
    int l = l0 + px;
    int pos = dir_pos(k, l);
    float v = g_xa[((size_t)(b * DIN + d) << 12) + pos];
    s_a[px][d] = (_Float16)v;
    g_xst[(((size_t)bk << 12) + l) * DIN + d] = v;
  }
  __syncthreads();

  // x_dbl = x_proj_w[k] (38x192) @ xs_tile  -> (16 x 48) per wave
  gemm_lds(&s_a[wid * 16][0], DIN, xpw + (size_t)k * (RR + 2 * NN) * DIN, DIN,
           RR + 2 * NN, &s_v[wid * 16][0], 48, 3, 6, nullptr, lane);
  __syncthreads();

  if (tid < 64) {
    int l = l0 + tid;
    size_t base = ((size_t)bk << 12) + l;
    float* xdbl = s_v[tid];
    for (int n = 0; n < NN; ++n) {
      g_Bs[base * NN + n] = xdbl[RR + n];
      g_Cs[base * NN + n] = xdbl[RR + NN + n];
    }
    const float* dw = dtw + (size_t)k * DIN * RR;
    const float* db = dtb + k * DIN;
    for (int d = 0; d < DIN; ++d) {
      float s = db[d];
#pragma unroll
      for (int r = 0; r < RR; ++r) s += xdbl[r] * dw[d * RR + r];
      float sp = (s > 20.f) ? s : log1pf(__expf(s));   // softplus
      g_dts[base * DIN + d] = sp;
    }
  }
}

// ======================================================================
// Kernel 4a: zero the y accumulator
// ======================================================================
__global__ void k_zero(float* __restrict__ p, long n) {
  size_t i = (size_t)blockIdx.x * 256 + threadIdx.x;
  if (i < (size_t)n) p[i] = 0.f;
}

// ======================================================================
// Kernel 4b: selective scan. block per (b,k): 192 threads, one per d,
// h[16] in registers, B/C broadcast via LDS, atomic scatter with the
// direction map (merges all 4 directions into g_y).
// ======================================================================
__global__ __launch_bounds__(192) void k_scan(
    const float* __restrict__ g_xst, const float* __restrict__ g_dts,
    const float* __restrict__ g_Bs, const float* __restrict__ g_Cs,
    const float* __restrict__ alog, const float* __restrict__ Ds,
    float* __restrict__ g_y) {
  __shared__ float sB[NN], sC[NN];
  int bk = blockIdx.x;
  int k = bk & 3, b = bk >> 2;
  int d = threadIdx.x;
  float A_[NN], h[NN];
#pragma unroll
  for (int n = 0; n < NN; ++n) {
    A_[n] = -__expf(alog[((size_t)(k * DIN + d)) * NN + n]);
    h[n] = 0.f;
  }
  float Dv = Ds[k * DIN + d];
  for (int l = 0; l < LL; ++l) {
    size_t base = ((size_t)bk << 12) + l;
    if (threadIdx.x < 32) {
      int n = threadIdx.x & 15;
      if (threadIdx.x < 16) sB[n] = g_Bs[base * NN + n];
      else                  sC[n] = g_Cs[base * NN + n];
    }
    __syncthreads();
    float dt  = g_dts[base * DIN + d];
    float x   = g_xst[base * DIN + d];
    float dtx = dt * x;
    float y   = Dv * x;
#pragma unroll
    for (int n = 0; n < NN; ++n) {
      float dA = __expf(dt * A_[n]);
      h[n] = dA * h[n] + dtx * sB[n];
      y += h[n] * sC[n];
    }
    atomicAdd(&g_y[((size_t)(b * DIN + d) << 12) + dir_pos(k, l)], y);
    __syncthreads();
  }
}

// ======================================================================
// Kernel 5: out_norm -> * SiLU(z) -> out_proj (WMMA) -> + x_sum residual
// ======================================================================
__global__ __launch_bounds__(128) void k_out(
    const float* __restrict__ g_y, const float* __restrict__ g_z,
    const float* __restrict__ onw, const float* __restrict__ onb,
    const float* __restrict__ wo, const float* __restrict__ g_xsum,
    float* __restrict__ out) {
  __shared__ alignas(32) float    s_v[64][DIN];
  __shared__ alignas(32) _Float16 s_a[64][DIN];
  int tid = threadIdx.x, lane = tid & 31, wid = tid >> 5;
  int pix0 = blockIdx.x * 64;
  __builtin_prefetch(wo, 0, 1);

  for (int idx = tid; idx < 64 * DIN; idx += 128) {
    int d = idx >> 6, px = idx & 63;
    int p = pix0 + px, b = p >> 12, hw = p & 4095;
    s_v[px][d] = g_y[((size_t)(b * DIN + d) << 12) + hw];
  }
  __syncthreads();

  if (tid < 64) {
    int p = pix0 + tid, b = p >> 12, hw = p & 4095;
    float* row = s_v[tid];
    ln_row(row, DIN, onw, onb);
    for (int d = 0; d < DIN; ++d) {
      float z = g_z[((size_t)(b * DIN + d) << 12) + hw];
      float sz = z / (1.f + __expf(-z));
      s_a[tid][d] = (_Float16)(row[d] * sz);
    }
  }
  __syncthreads();

  // out_proj (96x192): 6 N-tiles x 6 K-steps
  for (int t = 0; t < 6; ++t) {
    v8f acc;
#pragma unroll
    for (int r = 0; r < 8; ++r) acc[r] = 0.f;
    int c = t * 16 + (lane & 15);
    for (int ks = 0; ks < 6; ++ks) {
      v16h a  = afrag(&s_a[wid * 16][0], DIN, ks * 32, lane);
      v16h bf = bfrag_g(wo, DIN, c, ks * 32, lane);
      acc = wmma_acc(a, bf, acc);
    }
    int mo = (lane >> 4) * 8;
#pragma unroll
    for (int r = 0; r < 8; ++r) {
      int p = pix0 + wid * 16 + mo + r;
      int b = p >> 12, hw = p & 4095;
      size_t g = ((size_t)(b * CC + c) << 12) + hw;
      out[g] = acc[r] + g_xsum[g];
    }
  }
}

// ======================================================================
// launcher
// ======================================================================
extern "C" void kernel_launch(void* const* d_in, const int* in_sizes, int n_in,
                              void* d_out, int out_size, void* d_ws,
                              size_t ws_size, hipStream_t stream) {
  const float* x_deep   = (const float*)d_in[0];
  const float* x_shal   = (const float*)d_in[1];
  const float* conv1_w  = (const float*)d_in[2];
  const float* conv1_b  = (const float*)d_in[3];
  const float* ln1_w    = (const float*)d_in[4];
  const float* ln1_b    = (const float*)d_in[5];
  const float* conv2_w  = (const float*)d_in[6];
  const float* conv2_b  = (const float*)d_in[7];
  const float* ln2_w    = (const float*)d_in[8];
  const float* ln2_b    = (const float*)d_in[9];
  const float* norm_w   = (const float*)d_in[10];
  const float* norm_b   = (const float*)d_in[11];
  const float* in_proj  = (const float*)d_in[12];
  const float* conv_dw  = (const float*)d_in[13];
  const float* conv_dwb = (const float*)d_in[14];
  const float* x_proj   = (const float*)d_in[15];
  const float* dt_w     = (const float*)d_in[16];
  const float* dt_b     = (const float*)d_in[17];
  const float* A_logs   = (const float*)d_in[18];
  const float* Ds       = (const float*)d_in[19];
  const float* onorm_w  = (const float*)d_in[20];
  const float* onorm_b  = (const float*)d_in[21];
  const float* out_proj = (const float*)d_in[22];
  float* outp = (float*)d_out;

  float* ws = (float*)d_ws;
  size_t o = 0;
  float* g_xsum = ws + o; o += (size_t)BB * CC * LL;         // 1.57M
  float* g_xm   = ws + o; o += (size_t)BB * DIN * LL;        // 3.15M
  float* g_z    = ws + o; o += (size_t)BB * DIN * LL;
  float* g_xa   = ws + o; o += (size_t)BB * DIN * LL;
  float* g_xst  = ws + o; o += (size_t)BB * KK * LL * DIN;   // 12.6M
  float* g_dts  = ws + o; o += (size_t)BB * KK * LL * DIN;
  float* g_Bs   = ws + o; o += (size_t)BB * KK * LL * NN;
  float* g_Cs   = ws + o; o += (size_t)BB * KK * LL * NN;
  float* g_y    = ws + o; o += (size_t)BB * DIN * LL;

  k_pre<<<256, 128, 0, stream>>>(x_deep, x_shal, conv1_w, conv1_b, ln1_w, ln1_b,
                                 conv2_w, conv2_b, ln2_w, ln2_b, norm_w, norm_b,
                                 in_proj, g_xsum, g_xm, g_z);
  k_dwconv<<<(BB * DIN * LL + 255) / 256, 256, 0, stream>>>(g_xm, conv_dw,
                                                            conv_dwb, g_xa);
  k_proj<<<BB * KK * (LL / 64), 128, 0, stream>>>(g_xa, x_proj, dt_w, dt_b,
                                                  g_xst, g_dts, g_Bs, g_Cs);
  long ny = (long)BB * DIN * LL;
  k_zero<<<(int)((ny + 255) / 256), 256, 0, stream>>>(g_y, ny);
  k_scan<<<BB * KK, DIN, 0, stream>>>(g_xst, g_dts, g_Bs, g_Cs, A_logs, Ds, g_y);
  k_out<<<256, 128, 0, stream>>>(g_y, g_z, onorm_w, onorm_b, out_proj, g_xsum,
                                 outp);
}
